// MinGRUCell_48859547959729
// MI455X (gfx1250) — compile-verified, matrix operand order checked by reading, not run on hardware
//
#include <hip/hip_runtime.h>
#include <hip/hip_bf16.h>

typedef __attribute__((ext_vector_type(16))) __bf16 v16bf;
typedef __attribute__((ext_vector_type(8)))  float  v8f;
typedef int v4i_t __attribute__((vector_size(16)));   // matches builtin param type

#define AS1 __attribute__((address_space(1)))
#define AS3 __attribute__((address_space(3)))

#if defined(__has_builtin)
# if __has_builtin(__builtin_amdgcn_global_load_async_to_lds_b128)
#  define HAVE_ASYNC_LDS 1
# endif
#endif

#define B_    8
#define S_    4096
#define DIN_  768
#define H_    768
#define M_    (B_ * S_)        // 32768 rows of the fused GEMM
#define CHUNK 256
#define NCHUNK (S_ / CHUNK)    // 16

// GEMM tiling: 256 threads = 8 waves; tile 128(M) x 64(N), K-step 32 (bf16)
#define TM 128
#define TN 64
#define TK 32
#define LDSW 40                // LDS row stride in bf16 elems (80B: 16B-aligned segs)

// ---- 16B global->LDS copy: async DMA (ASYNCcnt) if available, sync fallback
__device__ __forceinline__ void copy16(const __bf16* g, __bf16* l) {
#ifdef HAVE_ASYNC_LDS
    __builtin_amdgcn_global_load_async_to_lds_b128(
        (AS1 v4i_t*)(g), (AS3 v4i_t*)(l), 0, 0);
#else
    *(uint4*)l = *(const uint4*)g;
#endif
}
__device__ __forceinline__ void wait_copies() {
#ifdef HAVE_ASYNC_LDS
    asm volatile("s_wait_asynccnt 0" ::: "memory");
#endif
}

// ---- pack two f32 into packed bf16 dword
__device__ __forceinline__ unsigned int pk_bf16(float lo, float hi) {
    __bf16 l = (__bf16)lo, h = (__bf16)hi;
    unsigned short ul = __builtin_bit_cast(unsigned short, l);
    unsigned short uh = __builtin_bit_cast(unsigned short, h);
    return (unsigned int)ul | ((unsigned int)uh << 16);
}

// -------------------------------------------------------------------------
// Kernel 0: streaming f32 -> bf16 (8 elems/thread, b128 in / b128 out)
// -------------------------------------------------------------------------
__global__ __launch_bounds__(256)
void cvt_bf16(const float* __restrict__ in, __bf16* __restrict__ out)
{
    const size_t i = ((size_t)blockIdx.x * 256 + threadIdx.x) * 8;
    float4 a = *(const float4*)(in + i);
    float4 b = *(const float4*)(in + i + 4);
    uint4 r;
    r.x = pk_bf16(a.x, a.y);
    r.y = pk_bf16(a.z, a.w);
    r.z = pk_bf16(b.x, b.y);
    r.w = pk_bf16(b.z, b.w);
    *(uint4*)(out + i) = r;
}

// -------------------------------------------------------------------------
// Kernel 1: fused dual GEMM (k = x*Wz^T + bz, p = x*Wh^T + bh) + epilogue.
// Double-buffered async LDS staging; 8x v_wmma_f32_16x16x32_bf16 per K-step.
// Writes linear-space scan operands:
//   coeff[m,n] = sigmoid(-k)       vals[m,n] = sigmoid(k) * g(p)
// with g(x) = x>=0 ? x+0.5 : sigmoid(x)  (== exp(log_g(x)))
// -------------------------------------------------------------------------
__global__ __launch_bounds__(256)
void mingru_proj(const __bf16* __restrict__ xb,
                 const __bf16* __restrict__ wzb,
                 const __bf16* __restrict__ whb,
                 const float* __restrict__ bz, const float* __restrict__ bh,
                 float* __restrict__ coeff,     // [M_, H_]  (d_out reused as temp)
                 float* __restrict__ vals)      // [M_, H_]  (workspace)
{
    __shared__ __align__(16) __bf16 sA [2][TM * LDSW];   // 2 x 10.0 KB
    __shared__ __align__(16) __bf16 sBz[2][TN * LDSW];   // 2 x  5.0 KB
    __shared__ __align__(16) __bf16 sBh[2][TN * LDSW];   // 2 x  5.0 KB

    const int tid   = threadIdx.x;
    const int lane  = tid & 31;
    const int wave  = tid >> 5;
    const int waveM = wave & 3;        // 4 waves down M  -> 32 rows each
    const int waveN = wave >> 2;       // 2 waves across N -> 32 cols each
    const int half  = lane >> 4;
    const int l16   = lane & 15;

    const int mTile = blockIdx.y * TM;
    const int nTile = blockIdx.x * TN;

    // stage one K-tile (buf) from global bf16 via 16B transfers
    auto stage = [&](int buf, int kk) {
        // x tile: 128 rows x 4 segs = 512 transfers, 2/thread
        int e = tid;
#pragma unroll
        for (int i = 0; i < 2; ++i, e += 256) {
            int row = e >> 2, seg = (e & 3) * 8;
            copy16(&xb[(size_t)(mTile + row) * DIN_ + kk + seg],
                   &sA[buf][row * LDSW + seg]);
        }
        // W tiles: 64 rows x 4 segs = 256 transfers each, 1/thread each
        int row = tid >> 2, seg = (tid & 3) * 8;
        copy16(&wzb[(size_t)(nTile + row) * DIN_ + kk + seg],
               &sBz[buf][row * LDSW + seg]);
        copy16(&whb[(size_t)(nTile + row) * DIN_ + kk + seg],
               &sBh[buf][row * LDSW + seg]);
    };

    v8f acc[2][2][2];
#pragma unroll
    for (int a = 0; a < 2; ++a)
#pragma unroll
        for (int b = 0; b < 2; ++b)
#pragma unroll
            for (int c = 0; c < 2; ++c)
                acc[a][b][c] = (v8f){0.f,0.f,0.f,0.f,0.f,0.f,0.f,0.f};

    stage(0, 0);
    wait_copies();
    __syncthreads();

    int buf = 0;
    for (int kk = 0; kk < DIN_; kk += TK) {
        if (kk + TK < DIN_) stage(buf ^ 1, kk + TK);   // overlap DMA with WMMA

        // ---- fragments per ISA 16-bit A 16x32 layout:
        // element e -> K = e + half*8 + (e>=8 ? 8 : 0)
        v16bf af[2], bzf[2], bhf[2];
#pragma unroll
        for (int sub = 0; sub < 2; ++sub) {
            const int rowA = waveM * 32 + sub * 16 + l16;   // A: lane%16 = M
            const int colB = waveN * 32 + sub * 16 + l16;   // B: lane%16 = N
#pragma unroll
            for (int e = 0; e < 16; ++e) {
                int k = e + half * 8 + (e >= 8 ? 8 : 0);
                af [sub][e] = sA [buf][rowA * LDSW + k];
                bzf[sub][e] = sBz[buf][colB * LDSW + k];
                bhf[sub][e] = sBh[buf][colB * LDSW + k];
            }
        }

#pragma unroll
        for (int sm = 0; sm < 2; ++sm)
#pragma unroll
            for (int sn = 0; sn < 2; ++sn) {
                acc[0][sm][sn] = __builtin_amdgcn_wmma_f32_16x16x32_bf16(
                    false, af[sm], false, bzf[sn], (short)0, acc[0][sm][sn], false, false);
                acc[1][sm][sn] = __builtin_amdgcn_wmma_f32_16x16x32_bf16(
                    false, af[sm], false, bhf[sn], (short)0, acc[1][sm][sn], false, false);
            }

        wait_copies();
        __syncthreads();
        buf ^= 1;
    }

    // ---- epilogue: C/D layout lane<16 -> M=r, lane>=16 -> M=r+8; N = lane%16
#pragma unroll
    for (int sm = 0; sm < 2; ++sm)
#pragma unroll
        for (int sn = 0; sn < 2; ++sn) {
            const int n    = nTile + waveN * 32 + sn * 16 + l16;
            const float bzv = bz[n], bhv = bh[n];
#pragma unroll
            for (int r = 0; r < 8; ++r) {
                const int m  = mTile + waveM * 32 + sm * 16 + half * 8 + r;
                const float kv = acc[0][sm][sn][r] + bzv;
                const float pv = acc[1][sm][sn][r] + bhv;
                const float ek = __expf(kv);
                const float c  = 1.0f / (1.0f + ek);        // sigmoid(-k)
                const float z  = ek * c;                    // sigmoid(k)
                const float g  = (pv >= 0.0f) ? (pv + 0.5f)
                                              : 1.0f / (1.0f + __expf(-pv));
                const size_t idx = (size_t)m * H_ + n;
                coeff[idx] = c;
                vals [idx] = z * g;
            }
        }
}

// -------------------------------------------------------------------------
// Kernel 2: per-chunk affine reduction  h_end = P*h_start + Q
// -------------------------------------------------------------------------
__global__ __launch_bounds__(256)
void mingru_chunk_reduce(const float* __restrict__ coeff,
                         const float* __restrict__ vals,
                         float* __restrict__ P, float* __restrict__ Q)
{
    const int h  = blockIdx.x * 256 + threadIdx.x;
    const int b  = blockIdx.y;
    const int ch = blockIdx.z;
    size_t base = ((size_t)(b * S_ + ch * CHUNK)) * H_ + h;
    float p = 1.0f, q = 0.0f;
    for (int t = 0; t < CHUNK; ++t) {
        float c = coeff[base];
        float w = vals [base];
        p *= c;
        q  = c * q + w;
        base += H_;
    }
    const int sidx = (b * H_ + h) * NCHUNK + ch;
    P[sidx] = p;
    Q[sidx] = q;
}

// -------------------------------------------------------------------------
// Kernel 3: chain NCHUNK affine maps per (b,h), seeded with g(h_0)
// -------------------------------------------------------------------------
__global__ __launch_bounds__(256)
void mingru_combine(const float* __restrict__ h0,
                    const float* __restrict__ P, const float* __restrict__ Q,
                    float* __restrict__ start)
{
    const int i = blockIdx.x * 256 + threadIdx.x;   // i = b*H + h
    if (i >= B_ * H_) return;
    const float hv = h0[i];
    float hs = (hv >= 0.0f) ? (hv + 0.5f) : 1.0f / (1.0f + __expf(-hv)); // g(h0)
#pragma unroll
    for (int ch = 0; ch < NCHUNK; ++ch) {
        start[i * NCHUNK + ch] = hs;
        hs = P[i * NCHUNK + ch] * hs + Q[i * NCHUNK + ch];
    }
}

// -------------------------------------------------------------------------
// Kernel 4: replay each chunk with correct incoming state; h overwrites the
// coeff temp in d_out (each element read once by same thread before write)
// -------------------------------------------------------------------------
__global__ __launch_bounds__(256)
void mingru_scan_out(const float* __restrict__ vals,
                     const float* __restrict__ start,
                     float* __restrict__ out)
{
    const int h  = blockIdx.x * 256 + threadIdx.x;
    const int b  = blockIdx.y;
    const int ch = blockIdx.z;
    float hv = start[(b * H_ + h) * NCHUNK + ch];
    size_t base = ((size_t)(b * S_ + ch * CHUNK)) * H_ + h;
    for (int t = 0; t < CHUNK; ++t) {
        const float c = out [base];
        const float w = vals[base];
        hv = c * hv + w;
        out[base] = hv;
        base += H_;
    }
}

// -------------------------------------------------------------------------
extern "C" void kernel_launch(void* const* d_in, const int* in_sizes, int n_in,
                              void* d_out, int out_size, void* d_ws, size_t ws_size,
                              hipStream_t stream)
{
    const float* x  = (const float*)d_in[0];
    const float* h0 = (const float*)d_in[1];
    const float* Wz = (const float*)d_in[2];
    const float* bz = (const float*)d_in[3];
    const float* Wh = (const float*)d_in[4];
    const float* bh = (const float*)d_in[5];
    float* out = (float*)d_out;

    // Workspace: vals f32[M*H] | xb bf16[M*DIN] | wzb,whb bf16[H*DIN] | P|Q|start
    float*  vals  = (float*)d_ws;
    __bf16* xb    = (__bf16*)(vals + (size_t)M_ * H_);
    __bf16* wzb   = xb  + (size_t)M_ * DIN_;
    __bf16* whb   = wzb + (size_t)H_ * DIN_;
    float*  P     = (float*)(whb + (size_t)H_ * DIN_);
    float*  Q     = P + (size_t)B_ * H_ * NCHUNK;
    float*  start = Q + (size_t)B_ * H_ * NCHUNK;

    // Phase 0: f32 -> bf16 copies (all sizes divisible by 8*256)
    cvt_bf16<<<(M_ * DIN_) / (8 * 256), 256, 0, stream>>>(x,  xb);
    cvt_bf16<<<(H_ * DIN_) / (8 * 256), 256, 0, stream>>>(Wz, wzb);
    cvt_bf16<<<(H_ * DIN_) / (8 * 256), 256, 0, stream>>>(Wh, whb);

    // Phase 1: projections + epilogue
    dim3 g1(H_ / TN, M_ / TM);           // (12, 256)
    mingru_proj<<<g1, 256, 0, stream>>>(xb, wzb, whb, bz, bh, out, vals);

    // Phase 2-4: chunked linear-recurrence scan
    dim3 g2(H_ / 256, B_, NCHUNK);       // (3, 8, 16)
    mingru_chunk_reduce<<<g2, 256, 0, stream>>>(out, vals, P, Q);
    mingru_combine<<<(B_ * H_ + 255) / 256, 256, 0, stream>>>(h0, P, Q, start);
    mingru_scan_out<<<g2, 256, 0, stream>>>(vals, start, out);
}
